// WindowAttention_27693949124663
// MI455X (gfx1250) — matrix operand order repaired and checked by reading, not compiled, not run on hardware
//
#include <hip/hip_runtime.h>
#include <hip/hip_bf16.h>

typedef __attribute__((ext_vector_type(16))) _Float16 v16h;
typedef __attribute__((ext_vector_type(8)))  _Float16 v8h;
typedef __attribute__((ext_vector_type(8)))  float    v8f;

union ABFrag { v16h v; v8h h[2]; };

#define WMMA_F16(A, B, C) \
  __builtin_amdgcn_wmma_f32_16x16x32_f16(false, (A), false, (B), (short)0, (C), false, false)

// ---------------------------------------------------------------------------
// Elementwise f32 -> f16 conversion (grid-stride)
// ---------------------------------------------------------------------------
__global__ void wa_f32_to_f16(const float* __restrict__ in,
                              _Float16* __restrict__ out, long long n) {
  long long i = (long long)blockIdx.x * blockDim.x + threadIdx.x;
  long long stride = (long long)gridDim.x * blockDim.x;
  for (; i < n; i += stride) out[i] = (_Float16)in[i];
}

// ---------------------------------------------------------------------------
// Precompute padded relative-position bias: biasmat[h][64][64], pad = -1e30
// so padded columns vanish in softmax.  grid = 12 (one block per head).
// ---------------------------------------------------------------------------
__global__ void wa_bias_fill(const float* __restrict__ bias_table, // (169, 12)
                             const int* __restrict__ rel_index,    // (49, 49)
                             float* __restrict__ biasmat) {        // (12, 64, 64)
  const int h = blockIdx.x;
  for (int idx = threadIdx.x; idx < 64 * 64; idx += blockDim.x) {
    int m = idx >> 6, n = idx & 63;
    float v = -1e30f;
    if (m < 49 && n < 49) v = bias_table[rel_index[m * 49 + n] * 12 + h];
    biasmat[h * 4096 + idx] = v;
  }
}

// ---------------------------------------------------------------------------
// WMMA GEMM:  Out[M,N] = A[M,K](f16,row) * Bt[N,K](f16,row)^T + bias[N]
// Block tile 64(M) x 128(N), 256 threads = 8 waves, each wave 32x32.
// K stepped by 32 through LDS; next K-tile prefetched (global_prefetch_b8)
// while current tile feeds the WMMAs.  M%64==0, N%128==0, K%32==0 assumed.
// ---------------------------------------------------------------------------
template <bool OUT16>
__global__ __launch_bounds__(256)
void wa_gemm_wmma(const _Float16* __restrict__ A,
                  const _Float16* __restrict__ Bt,
                  const float* __restrict__ bias,
                  void* __restrict__ Out,
                  int M, int N, int K) {
  __shared__ __align__(16) _Float16 As[64 * 32];
  __shared__ __align__(16) _Float16 Bs[128 * 32];

  const int tid  = threadIdx.x;
  const int lane = tid & 31;
  const int wid  = tid >> 5;
  const int wm   = (wid >> 2) * 32;   // 0 or 32
  const int wn   = (wid & 3) * 32;    // 0,32,64,96
  const int hi   = lane >> 4;
  const int lm   = lane & 15;
  const long long M0 = (long long)blockIdx.x * 64;
  const long long N0 = (long long)blockIdx.y * 128;

  const int arow = tid >> 2, acg = tid & 3;   // A stage: 64 rows x (4 x 8 halves)
  const int brow = tid >> 1, bcg = tid & 1;   // B stage: 128 rows x (2 x 16 halves)
  const _Float16* aSrc = A  + (M0 + arow) * (long long)K + acg * 8;
  const _Float16* bSrc = Bt + (N0 + brow) * (long long)K + bcg * 16;

  v8f acc[2][2] = {};

  for (int k0 = 0; k0 < K; k0 += 32) {
    *(v8h*)&As[arow * 32 + acg * 8] = *(const v8h*)(aSrc + k0);
    {
      const v8h* src = (const v8h*)(bSrc + k0);
      *(v8h*)&Bs[brow * 32 + bcg * 16]     = src[0];
      *(v8h*)&Bs[brow * 32 + bcg * 16 + 8] = src[1];
    }
    if (k0 + 32 < K) {   // overlap next tile's HBM fetch with this tile's WMMAs
      __builtin_prefetch(aSrc + k0 + 32, 0, 0);
      __builtin_prefetch(bSrc + k0 + 32, 0, 0);
    }
    __syncthreads();

    ABFrag afr[2], bfr[2];
    for (int i = 0; i < 2; ++i) {
      int row = wm + i * 16 + lm;
      afr[i].h[0] = *(const v8h*)&As[row * 32 + 8 * hi];       // K = 8*hi + j
      afr[i].h[1] = *(const v8h*)&As[row * 32 + 16 + 8 * hi];  // K = 16 + 8*hi + j
    }
    for (int j = 0; j < 2; ++j) {
      int row = wn + j * 16 + lm;
      bfr[j].h[0] = *(const v8h*)&Bs[row * 32 + 8 * hi];       // K = 8*hi + j
      bfr[j].h[1] = *(const v8h*)&Bs[row * 32 + 16 + 8 * hi];
    }
    for (int i = 0; i < 2; ++i)
      for (int j = 0; j < 2; ++j)
        acc[i][j] = WMMA_F16(afr[i].v, bfr[j].v, acc[i][j]);
    __syncthreads();
  }

  for (int i = 0; i < 2; ++i)
    for (int j = 0; j < 2; ++j) {
      long long col = N0 + wn + j * 16 + lm;
      float bv = bias[col];
      for (int r = 0; r < 8; ++r) {
        long long rowg = M0 + wm + i * 16 + r + 8 * hi;  // c-frag: m = r + 8*hi
        float v = acc[i][j][r] + bv;
        if (OUT16) ((_Float16*)Out)[rowg * N + col] = (_Float16)v;
        else       ((float*)Out)[rowg * N + col]    = v;
      }
    }
}

// ---------------------------------------------------------------------------
// Fused window attention: one 128-thread block (4 waves) per (window b, head h).
// Wave w owns row-stripe mi = w (16 rows of S / P / O), so 4 waves share the
// 36 KB LDS footprint -> ~32 resident waves per WGP instead of 8.
// qkvh: (B*49, 1152) f16 with q @ +0, k @ +384, v @ +768 (head h slice = h*32).
// Writes attn output (B*49, 384) f16.
// ---------------------------------------------------------------------------
__global__ __launch_bounds__(128)
void wa_attn_kernel(const _Float16* __restrict__ qkvh,
                    const float* __restrict__ biasmat,   // (12, 64, 64)
                    _Float16* __restrict__ attnh) {
  __shared__ __align__(16) _Float16 Qs[64 * 32];
  __shared__ __align__(16) _Float16 Ks[64 * 32];
  __shared__ __align__(16) _Float16 Vt[32 * 64];   // transposed: [d][token]
  __shared__ __align__(16) float    Ss[64 * 64];
  __shared__ __align__(16) _Float16 Ps[64 * 64];

  const int b   = blockIdx.x;
  const int h   = blockIdx.y;
  const int tid = threadIdx.x;
  const int l   = tid & 31;
  const int mi  = tid >> 5;            // wave id == row-stripe (wave-uniform)
  const int hi  = l >> 4, lm = l & 15;
  const float scale = 0.17677669529663687f;  // 1/sqrt(32)

  // Stage Q, K, V^T: one row per thread for tid < 64 (pad rows 49..63 zero)
  if (tid < 64) {
    const int r = tid;
    if (r < 49) {
      const _Float16* base = qkvh + ((long long)(b * 49 + r)) * 1152 + h * 32;
      const v8h* q4 = (const v8h*)(base);
      const v8h* k4 = (const v8h*)(base + 384);
      const v8h* v4 = (const v8h*)(base + 768);
      for (int p = 0; p < 4; ++p) {
        *(v8h*)&Qs[r * 32 + p * 8] = q4[p];
        *(v8h*)&Ks[r * 32 + p * 8] = k4[p];
      }
      v8h v0 = v4[0], v1 = v4[1], v2 = v4[2], v3 = v4[3];
      for (int d = 0; d < 8; ++d) {
        Vt[(d)      * 64 + r] = v0[d];
        Vt[(d + 8)  * 64 + r] = v1[d];
        Vt[(d + 16) * 64 + r] = v2[d];
        Vt[(d + 24) * 64 + r] = v3[d];
      }
    } else {
      v8h z = {};
      for (int p = 0; p < 4; ++p) {
        *(v8h*)&Qs[r * 32 + p * 8] = z;
        *(v8h*)&Ks[r * 32 + p * 8] = z;
      }
      for (int d = 0; d < 32; ++d) Vt[d * 64 + r] = (_Float16)0.0f;
    }
  }
  __syncthreads();

  // S = scale * Q K^T + bias : this wave's 16-row stripe (4 WMMAs, K=32)
  {
    const float* bm = biasmat + (long long)h * 4096;
    ABFrag a;
    int ar = mi * 16 + lm;
    a.h[0] = *(const v8h*)&Qs[ar * 32 + 8 * hi];
    a.h[1] = *(const v8h*)&Qs[ar * 32 + 16 + 8 * hi];
    for (int nj = 0; nj < 4; ++nj) {
      ABFrag bf;
      int br = nj * 16 + lm;                           // token index (B col n)
      bf.h[0] = *(const v8h*)&Ks[br * 32 + 8 * hi];    // K-dim = 8*hi + j
      bf.h[1] = *(const v8h*)&Ks[br * 32 + 16 + 8 * hi];
      v8f c = {};
      c = WMMA_F16(a.v, bf.v, c);
      for (int r = 0; r < 8; ++r) {
        int m = mi * 16 + r + 8 * hi, n = nj * 16 + lm;
        Ss[m * 64 + n] = c[r] * scale + bm[m * 64 + n];
      }
    }
  }
  __syncthreads();

  // Row softmax over 64 cols (padded cols carry -1e30 -> exp == 0)
  if (tid < 64) {
    const int r = tid;
    float mx = -3.4e38f;
    for (int c2 = 0; c2 < 64; ++c2) mx = fmaxf(mx, Ss[r * 64 + c2]);
    float sum = 0.f;
    for (int c2 = 0; c2 < 64; ++c2) {
      float e = __expf(Ss[r * 64 + c2] - mx);
      Ss[r * 64 + c2] = e;
      sum += e;
    }
    float inv = 1.f / sum;
    for (int c2 = 0; c2 < 64; ++c2)
      Ps[r * 64 + c2] = (_Float16)(Ss[r * 64 + c2] * inv);
  }
  __syncthreads();

  // O = P @ V : this wave's stripe (2 n-tiles x 2 k-steps of 32 -> 4 WMMAs)
  for (int nj = 0; nj < 2; ++nj) {
    v8f c = {};
    for (int kk = 0; kk < 2; ++kk) {
      ABFrag a, bf;
      int ar = mi * 16 + lm;
      a.h[0] = *(const v8h*)&Ps[ar * 64 + kk * 32 + 8 * hi];
      a.h[1] = *(const v8h*)&Ps[ar * 64 + kk * 32 + 16 + 8 * hi];
      int br = nj * 16 + lm;  // output d index (row of Vt)
      bf.h[0] = *(const v8h*)&Vt[br * 64 + kk * 32 + 8 * hi];
      bf.h[1] = *(const v8h*)&Vt[br * 64 + kk * 32 + 16 + 8 * hi];
      c = WMMA_F16(a.v, bf.v, c);
    }
    for (int r = 0; r < 8; ++r) {
      int m = mi * 16 + r + 8 * hi;
      if (m < 49)
        attnh[((long long)(b * 49 + m)) * 384 + h * 32 + nj * 16 + lm] = (_Float16)c[r];
    }
  }
}

// ---------------------------------------------------------------------------
// Host launcher
// ---------------------------------------------------------------------------
extern "C" void kernel_launch(void* const* d_in, const int* in_sizes, int n_in,
                              void* d_out, int out_size, void* d_ws, size_t ws_size,
                              hipStream_t stream) {
  const float* x          = (const float*)d_in[0];  // (4096, 49, 384)
  const float* qkv_w      = (const float*)d_in[1];  // (1152, 384)
  const float* qkv_b      = (const float*)d_in[2];  // (1152)
  const float* proj_w     = (const float*)d_in[3];  // (384, 384)
  const float* proj_b     = (const float*)d_in[4];  // (384)
  const float* bias_table = (const float*)d_in[5];  // (169, 12)
  const int*   rel_index  = (const int*)d_in[6];    // (49, 49)
  float* out = (float*)d_out;

  const long long Mrows = 4096LL * 49;     // 200704
  const long long NX    = Mrows * 384;     // 77,070,336 elements

  // Workspace layout (all chunk sizes multiples of 256 bytes)
  char* ws = (char*)d_ws;
  size_t off = 0;
  _Float16* xh     = (_Float16*)(ws + off); off += (size_t)NX * 2;          // 154.1 MB
  _Float16* wqkvh  = (_Float16*)(ws + off); off += 1152ull * 384 * 2;       // 884 KB
  _Float16* wprojh = (_Float16*)(ws + off); off += 384ull * 384 * 2;        // 295 KB
  float*    biasm  = (float*)   (ws + off); off += 12ull * 64 * 64 * 4;     // 196 KB
  _Float16* qkvh   = (_Float16*)(ws + off); off += (size_t)Mrows * 1152 * 2;// 462.4 MB
  _Float16* attnh  = (_Float16*)(ws + off); off += (size_t)NX * 2;          // 154.1 MB

  // 1) precision conversion + bias precompute
  wa_f32_to_f16<<<4096, 256, 0, stream>>>(x, xh, NX);
  wa_f32_to_f16<<<432, 256, 0, stream>>>(qkv_w, wqkvh, 1152LL * 384);
  wa_f32_to_f16<<<144, 256, 0, stream>>>(proj_w, wprojh, 384LL * 384);
  wa_bias_fill<<<12, 256, 0, stream>>>(bias_table, rel_index, biasm);

  // 2) QKV projection: (200704 x 384) x (1152 x 384)^T -> f16
  wa_gemm_wmma<true><<<dim3(3136, 9), 256, 0, stream>>>(
      xh, wqkvh, qkv_b, (void*)qkvh, (int)Mrows, 1152, 384);

  // 3) fused window attention, 4 waves per (window, head)
  wa_attn_kernel<<<dim3(4096, 12), 128, 0, stream>>>(qkvh, biasm, attnh);

  // 4) output projection: (200704 x 384) x (384 x 384)^T -> f32 out
  wa_gemm_wmma<false><<<dim3(3136, 3), 256, 0, stream>>>(
      attnh, wprojh, proj_b, (void*)out, (int)Mrows, 384, 384);
}
